// ProtoAttn1_89043261981323
// MI455X (gfx1250) — matrix operand order repaired and checked by reading, not compiled, not run on hardware
//
#include <hip/hip_runtime.h>

// ---------------------------------------------------------------------------
// Two-stage prototype attention for MI455X (gfx1250, wave32, WMMA + TDM).
// GEMMs: v_wmma_f32_16x16x32_bf16 with register double-buffering; A-tiles
// staged to LDS via tensor_load_to_lds (TDM) where they fit.
// ---------------------------------------------------------------------------

typedef __bf16 bf16_t;
typedef __attribute__((ext_vector_type(16))) __bf16 v16bf;
typedef __attribute__((ext_vector_type(8)))  __bf16 v8bf;
typedef __attribute__((ext_vector_type(8)))  float  v8f;
typedef unsigned int u32x4 __attribute__((ext_vector_type(4)));
typedef int          i32x8 __attribute__((ext_vector_type(8)));
typedef int          i32x4 __attribute__((ext_vector_type(4)));

static constexpr int  B_ = 16, T_ = 4096, H_ = 256, P_ = 64;
static constexpr long BT_ = (long)B_ * T_;   // 65536

// ---------------- fragment helpers (layouts per CDNA5 ISA 7.12.2) ----------

__device__ __forceinline__ v16bf make_v16(v8bf lo, v8bf hi) {
  v16bf r;
#pragma unroll
  for (int i = 0; i < 8; ++i) { r[i] = lo[i]; r[i + 8] = hi[i]; }
  return r;
}

// A fragment: 16x32 bf16 tile, source row-major [m][k] (ptr at m0,k0).
__device__ __forceinline__ v16bf load_a_frag(const bf16_t* __restrict__ a,
                                             int lda, int lane) {
  const bf16_t* p = a + (long)(lane & 15) * lda + ((lane >> 4) << 3);
  v8bf lo = *(const v8bf*)p;
  v8bf hi = *(const v8bf*)(p + 16);
  return make_v16(lo, hi);
}

// B fragment: 32x16 (KxN) tile, source row-major [n][k] (ptr at n0,k0).
__device__ __forceinline__ v16bf load_b_frag(const bf16_t* __restrict__ w,
                                             int ldw, int lane) {
  const bf16_t* p = w + (long)(lane & 15) * ldw + ((lane >> 4) << 3);
  v8bf lo = *(const v8bf*)p;
  v8bf hi = *(const v8bf*)(p + 16);
  return make_v16(lo, hi);
}

__device__ __forceinline__ v8f wmma_bf16(v16bf a, v16bf b, v8f c) {
  return __builtin_amdgcn_wmma_f32_16x16x32_bf16(false, a, false, b,
                                                 (short)0, c, false, false);
}

template <int NF>
__device__ __forceinline__ void zero_acc(v8f acc[NF]) {
#pragma unroll
  for (int f = 0; f < NF; ++f)
#pragma unroll
    for (int i = 0; i < 8; ++i) acc[f][i] = 0.0f;
}

// Double-buffered K pipeline (requires K % 64 == 0, K >= 64).
template <int NF>
__device__ __forceinline__ void gemm_pipe(const bf16_t* __restrict__ arow, int lda,
                                          const bf16_t* __restrict__ wrow, int ldw,
                                          int K, int lane, v8f acc[NF]) {
  v16bf a0, a1, b0[NF], b1[NF];
  a0 = load_a_frag(arow, lda, lane);
#pragma unroll
  for (int f = 0; f < NF; ++f) b0[f] = load_b_frag(wrow + (long)f * 16 * ldw, ldw, lane);
  for (int k0 = 0; k0 + 64 <= K; k0 += 64) {
    __builtin_prefetch(wrow + k0 + 128, 0, 1);
    a1 = load_a_frag(arow + k0 + 32, lda, lane);
#pragma unroll
    for (int f = 0; f < NF; ++f)
      b1[f] = load_b_frag(wrow + (long)f * 16 * ldw + k0 + 32, ldw, lane);
#pragma unroll
    for (int f = 0; f < NF; ++f) acc[f] = wmma_bf16(a0, b0[f], acc[f]);
    if (k0 + 64 < K) {
      a0 = load_a_frag(arow + k0 + 64, lda, lane);
#pragma unroll
      for (int f = 0; f < NF; ++f)
        b0[f] = load_b_frag(wrow + (long)f * 16 * ldw + k0 + 64, ldw, lane);
    }
#pragma unroll
    for (int f = 0; f < NF; ++f) acc[f] = wmma_bf16(a1, b1[f], acc[f]);
  }
}

// ---------------- TDM staging of a 64x256 bf16 tile into LDS ---------------
// Descriptor per CDNA5 ISA 8.3/8.4: 2D tensor, data_size=2B, tile 256x64,
// contiguous (stride 256). Issued by wave 0 only; EXEC ignored by TDM.

#if __has_builtin(__builtin_amdgcn_tensor_load_to_lds)
#define ATHENA_HAVE_TDM 1
#else
#define ATHENA_HAVE_TDM 0
#endif

__device__ __forceinline__ void stage_a_tile(bf16_t* __restrict__ dstLDS,
                                             const bf16_t* __restrict__ src,
                                             int tid) {
#if ATHENA_HAVE_TDM
  if (tid < 32) {
    unsigned long long ga = (unsigned long long)(const void*)src;
    u32x4 g0;
    g0[0] = 1u;                                            // count=1, user desc
    g0[1] = 0u;                                            // lds_addr (tile at 0)
    g0[2] = (unsigned)(ga & 0xffffffffull);                // global_addr lo
    g0[3] = (unsigned)((ga >> 32) & 0x01ffffffull) | (2u << 30);  // hi | type=2
    i32x8 g1;
    g1[0] = (int)(1u << 16);          // workgroup_mask=0, data_size=1 (2 bytes)
    g1[1] = (int)(256u << 16);        // tensor_dim0[15:0]=256 at bits 63:48
    g1[2] = (int)(64u << 16);         // tensor_dim0 hi=0 | tensor_dim1[15:0]=64
    g1[3] = (int)(256u << 16);        // tensor_dim1 hi=0 | tile_dim0=256
    g1[4] = (int)64;                  // tile_dim1=64, tile_dim2=0
    g1[5] = (int)256;                 // tensor_dim0_stride lo32 = 256
    g1[6] = 0;                        // stride hi | tensor_dim1_stride lo
    g1[7] = 0;
    i32x4 z4 = {0, 0, 0, 0};
    i32x8 z8 = {0, 0, 0, 0, 0, 0, 0, 0};
    __builtin_amdgcn_tensor_load_to_lds(g0, g1, z4, z4, z8, 0);
    __builtin_amdgcn_s_wait_tensorcnt(0);
  }
  __syncthreads();
#else
  for (int i = tid; i < (64 * 256) / 8; i += 256)
    ((v8bf*)dstLDS)[i] = ((const v8bf*)src)[i];
  __syncthreads();
#endif
}

// ---------------- kernels --------------------------------------------------

__global__ __launch_bounds__(256) void cvt_kernel(const float* __restrict__ s,
                                                  bf16_t* __restrict__ d, long n) {
  long i = (long)blockIdx.x * blockDim.x + threadIdx.x;
  long stride = (long)gridDim.x * blockDim.x;
  for (; i < n; i += stride) d[i] = (bf16_t)s[i];
}

// Y[M][256] = X[M][256] * W[256][256]^T + bias.  Block tile 64x256, 8 waves.
// A tile staged into LDS via TDM. Optional transposed bf16 store.
__global__ __launch_bounds__(256)
void gemm_proj_kernel(const bf16_t* __restrict__ X, const bf16_t* __restrict__ W,
                      const float* __restrict__ bias,
                      bf16_t* __restrict__ Y, bf16_t* __restrict__ Yt,
                      int rows_shift) {
  __shared__ bf16_t tileA[64 * 256];                       // 32 KiB
  const int tid = threadIdx.x;
  const int lane = tid & 31;
  const int w = tid >> 5;
  const long row0 = (long)blockIdx.x * 64;
  stage_a_tile(tileA, X + row0 * H_, tid);

  const int mstrip = (w & 3) * 16;
  const int ncol0 = (w >> 2) * 128;
  v8f acc[8];
  zero_acc<8>(acc);
  gemm_pipe<8>(tileA + mstrip * H_, H_, W + (long)ncol0 * H_, H_, H_, lane, acc);

  const int n_base = ncol0 + (lane & 15);
  const long m_base = row0 + mstrip + ((lane >> 4) << 3);
  const long tmask = (1L << rows_shift) - 1;
#pragma unroll
  for (int f = 0; f < 8; ++f) {
    const int n = n_base + f * 16;
    const float bv = bias[n];
#pragma unroll
    for (int r = 0; r < 8; ++r) {
      const long m = m_base + r;
      const float v = acc[f][r] + bv;
      if (Y) Y[m * 256 + n] = (bf16_t)v;
      if (Yt) {
        const long bb = m >> rows_shift, tt = m & tmask;
        Yt[((bb * 256 + n) << rows_shift) + tt] = (bf16_t)v;
      }
    }
  }
}

// scores1[b][p][t] = (q1 . k1[b,t]) * rsqrt(len_b), masked (t>=len -> -1e9).
__global__ __launch_bounds__(256)
void scores1_kernel(const bf16_t* __restrict__ q1, const bf16_t* __restrict__ k1,
                    const float* __restrict__ lens, float* __restrict__ S) {
  __shared__ bf16_t tileQ[64 * 256];                       // full q1, 32 KiB
  const int tid = threadIdx.x;
  const int lane = tid & 31;
  const int w = tid >> 5;
  const int b = blockIdx.y;
  stage_a_tile(tileQ, q1, tid);

  const int p0 = (w & 3) * 16;
  const int t0 = blockIdx.x * 256 + (w >> 2) * 128;
  v8f acc[8];
  zero_acc<8>(acc);
  gemm_pipe<8>(tileQ + p0 * H_, H_,
               k1 + ((long)b * T_ + t0) * H_, H_, H_, lane, acc);

  const float len = lens[b];
  const float scale = rsqrtf(len);
  const int ilen = (int)len;
  const int t_base = t0 + (lane & 15);
  const int p_base = p0 + ((lane >> 4) << 3);
#pragma unroll
  for (int f = 0; f < 8; ++f) {
    const int t = t_base + f * 16;
    const bool masked = (t >= ilen);
#pragma unroll
    for (int r = 0; r < 8; ++r) {
      float v = acc[f][r] * scale;
      if (masked) v = -1e9f;
      S[((long)b * P_ + p_base + r) * T_ + t] = v;
    }
  }
}

// softmax over T (4096), one wave per (b,p) row.
__global__ __launch_bounds__(256)
void softmax1_kernel(const float* __restrict__ S, bf16_t* __restrict__ A) {
  const int w = threadIdx.x >> 5, lane = threadIdx.x & 31;
  const long row = (long)blockIdx.x * 8 + w;
  const float* src = S + row * T_;
  float m = -3.0e38f, s = 0.0f;
  for (int t = lane; t < T_; t += 32) {
    const float x = src[t];
    if (x > m) { s *= __expf(m - x); m = x; }
    s += __expf(x - m);
  }
#pragma unroll
  for (int off = 16; off; off >>= 1) {
    const float mo = __shfl_xor(m, off, 32);
    const float so = __shfl_xor(s, off, 32);
    const float mn = fmaxf(m, mo);
    s = s * __expf(m - mn) + so * __expf(mo - mn);
    m = mn;
  }
  const float inv = 1.0f / s;
  bf16_t* dst = A + row * T_;
  for (int t = lane; t < T_; t += 32)
    dst[t] = (bf16_t)(__expf(src[t] - m) * inv);
}

// pf[b][p][h] = attn1[b] (64x4096) x v1t[b] ([h][t], ld=T).  grid = B.
__global__ __launch_bounds__(256)
void pf_kernel(const bf16_t* __restrict__ attn1, const bf16_t* __restrict__ v1t,
               bf16_t* __restrict__ pfb) {
  const int lane = threadIdx.x & 31, w = threadIdx.x >> 5;
  const int b = blockIdx.x;
  const int p0 = (w & 3) * 16;
  const int h0 = (w >> 2) * 128;
  v8f acc[8];
  zero_acc<8>(acc);
  gemm_pipe<8>(attn1 + ((long)b * P_ + p0) * T_, T_,
               v1t + ((long)b * H_ + h0) * T_, T_, T_, lane, acc);
  const int h_base = h0 + (lane & 15);
  const int p_base = p0 + ((lane >> 4) << 3);
#pragma unroll
  for (int f = 0; f < 8; ++f)
#pragma unroll
    for (int r = 0; r < 8; ++r)
      pfb[((long)b * P_ + p_base + r) * H_ + h_base + f * 16] = (bf16_t)acc[f][r];
}

// scores2[m][p] = (q2[m] . k2[b,p]) / 8.  Block tile 128M x 64N.
__global__ __launch_bounds__(256)
void scores2_kernel(const bf16_t* __restrict__ q2, const bf16_t* __restrict__ k2,
                    float* __restrict__ S) {
  const int lane = threadIdx.x & 31, w = threadIdx.x >> 5;
  const long row0 = (long)blockIdx.x * 128 + w * 16;
  const int b = (int)(row0 >> 12);
  v8f acc[4];
  zero_acc<4>(acc);
  gemm_pipe<4>(q2 + row0 * H_, H_, k2 + (long)b * P_ * H_, H_, H_, lane, acc);
  const int p_base = lane & 15;
  const long m_base = row0 + ((lane >> 4) << 3);
#pragma unroll
  for (int f = 0; f < 4; ++f)
#pragma unroll
    for (int r = 0; r < 8; ++r)
      S[(m_base + r) * P_ + p_base + f * 16] = acc[f][r] * 0.125f;
}

// softmax over P=64, one wave per row.
__global__ __launch_bounds__(256)
void softmax2_kernel(const float* __restrict__ S, bf16_t* __restrict__ A) {
  const int w = threadIdx.x >> 5, lane = threadIdx.x & 31;
  const long row = (long)blockIdx.x * 8 + w;
  const float* src = S + row * P_;
  const float x0 = src[lane], x1 = src[lane + 32];
  float m = fmaxf(x0, x1);
#pragma unroll
  for (int off = 16; off; off >>= 1) m = fmaxf(m, __shfl_xor(m, off, 32));
  const float e0 = __expf(x0 - m), e1 = __expf(x1 - m);
  float s = e0 + e1;
#pragma unroll
  for (int off = 16; off; off >>= 1) s += __shfl_xor(s, off, 32);
  const float inv = 1.0f / s;
  bf16_t* dst = A + row * P_;
  dst[lane] = (bf16_t)(e0 * inv);
  dst[lane + 32] = (bf16_t)(e1 * inv);
}

// out[m][h] = attn2[b] (Tx64) x v2t[b] ([h][p], ld=P). Block 64M x 256N.
__global__ __launch_bounds__(256)
void out_kernel(const bf16_t* __restrict__ attn2, const bf16_t* __restrict__ v2t,
                float* __restrict__ out) {
  const int lane = threadIdx.x & 31, w = threadIdx.x >> 5;
  const long row0 = (long)blockIdx.x * 64 + (w & 3) * 16;
  const int h0 = (w >> 2) * 128;
  const int b = (int)(row0 >> 12);
  v8f acc[8];
  zero_acc<8>(acc);
  gemm_pipe<8>(attn2 + row0 * P_, P_,
               v2t + ((long)b * H_ + h0) * P_, P_, P_, lane, acc);
  const int h_base = h0 + (lane & 15);
  const long m_base = row0 + ((lane >> 4) << 3);
#pragma unroll
  for (int f = 0; f < 8; ++f)
#pragma unroll
    for (int r = 0; r < 8; ++r)
      out[(m_base + r) * H_ + h_base + f * 16] = acc[f][r];
}

// ---------------- workspace layout (bytes) ---------------------------------
static constexpr size_t SZ_HSB = (size_t)BT_ * H_ * 2;          // 32 MiB
static constexpr size_t SZ_SC  = (size_t)B_ * P_ * T_ * 4;      // 16 MiB
static constexpr size_t SZ_ATT = (size_t)B_ * P_ * T_ * 2;      // 8 MiB
static constexpr size_t SZ_W   = (size_t)H_ * H_ * 2;
static constexpr size_t SZ_PH  = (size_t)B_ * P_ * H_ * 2;

static constexpr size_t OFF_HSB = 0;
static constexpr size_t OFF_K1B = OFF_HSB + SZ_HSB;
static constexpr size_t OFF_V1T = OFF_K1B + SZ_HSB;
static constexpr size_t OFF_Q2B = OFF_V1T + SZ_HSB;
static constexpr size_t OFF_SC  = OFF_Q2B + SZ_HSB;   // scores1, reused as scores2
static constexpr size_t OFF_ATT = OFF_SC + SZ_SC;     // attn1 bf16, reused as attn2
static constexpr size_t OFF_PRB = OFF_ATT + SZ_ATT;
static constexpr size_t OFF_Q1B = OFF_PRB + (size_t)P_ * H_ * 2;
static constexpr size_t OFF_WB  = OFF_Q1B + (size_t)P_ * H_ * 2;
static constexpr size_t OFF_PFB = OFF_WB + 6 * SZ_W;
static constexpr size_t OFF_K2B = OFF_PFB + SZ_PH;
static constexpr size_t OFF_V2T = OFF_K2B + SZ_PH;

extern "C" void kernel_launch(void* const* d_in, const int* in_sizes, int n_in,
                              void* d_out, int out_size, void* d_ws, size_t ws_size,
                              hipStream_t stream) {
  const float* hs    = (const float*)d_in[0];
  const float* proto = (const float*)d_in[1];
  const float* lens  = (const float*)d_in[2];
  // d_in[3] (bool mask) unused: mask recomputed from lens.
  const float* Ww[6] = {(const float*)d_in[4],  (const float*)d_in[6],
                        (const float*)d_in[8],  (const float*)d_in[10],
                        (const float*)d_in[12], (const float*)d_in[14]};
  const float* Wb[6] = {(const float*)d_in[5],  (const float*)d_in[7],
                        (const float*)d_in[9],  (const float*)d_in[11],
                        (const float*)d_in[13], (const float*)d_in[15]};

  char* ws = (char*)d_ws;
  bf16_t* hsb  = (bf16_t*)(ws + OFF_HSB);
  bf16_t* k1b  = (bf16_t*)(ws + OFF_K1B);
  bf16_t* v1t  = (bf16_t*)(ws + OFF_V1T);
  bf16_t* q2b  = (bf16_t*)(ws + OFF_Q2B);
  float*  sc   = (float*)(ws + OFF_SC);
  bf16_t* attb = (bf16_t*)(ws + OFF_ATT);
  bf16_t* prb  = (bf16_t*)(ws + OFF_PRB);
  bf16_t* q1b  = (bf16_t*)(ws + OFF_Q1B);
  bf16_t* wb   = (bf16_t*)(ws + OFF_WB);
  bf16_t* pfb  = (bf16_t*)(ws + OFF_PFB);
  bf16_t* k2b  = (bf16_t*)(ws + OFF_K2B);
  bf16_t* v2t  = (bf16_t*)(ws + OFF_V2T);

  // casts
  cvt_kernel<<<2048, 256, 0, stream>>>(hs, hsb, BT_ * H_);
  cvt_kernel<<<64, 256, 0, stream>>>(proto, prb, (long)P_ * H_);
  for (int i = 0; i < 6; ++i)
    cvt_kernel<<<64, 256, 0, stream>>>(Ww[i], wb + (size_t)i * H_ * H_, (long)H_ * H_);

  // stage-1 projections
  gemm_proj_kernel<<<1, 256, 0, stream>>>(prb, wb + 0 * (size_t)H_ * H_, Wb[0], q1b, nullptr, 0);
  gemm_proj_kernel<<<BT_ / 64, 256, 0, stream>>>(hsb, wb + 1 * (size_t)H_ * H_, Wb[1], k1b, nullptr, 0);
  gemm_proj_kernel<<<BT_ / 64, 256, 0, stream>>>(hsb, wb + 2 * (size_t)H_ * H_, Wb[2], nullptr, v1t, 12);
  gemm_proj_kernel<<<BT_ / 64, 256, 0, stream>>>(hsb, wb + 3 * (size_t)H_ * H_, Wb[3], q2b, nullptr, 0);

  // stage-1 attention
  scores1_kernel<<<dim3(T_ / 256, B_), 256, 0, stream>>>(q1b, k1b, lens, sc);
  softmax1_kernel<<<(B_ * P_) / 8, 256, 0, stream>>>(sc, attb);
  pf_kernel<<<B_, 256, 0, stream>>>(attb, v1t, pfb);

  // stage-2 projections (M = B*P = 1024)
  gemm_proj_kernel<<<(B_ * P_) / 64, 256, 0, stream>>>(pfb, wb + 4 * (size_t)H_ * H_, Wb[4], k2b, nullptr, 0);
  gemm_proj_kernel<<<(B_ * P_) / 64, 256, 0, stream>>>(pfb, wb + 5 * (size_t)H_ * H_, Wb[5], nullptr, v2t, 6);

  // stage-2 attention (reuse sc / attb buffers)
  scores2_kernel<<<BT_ / 128, 256, 0, stream>>>(q2b, k2b, sc);
  softmax2_kernel<<<BT_ / 8, 256, 0, stream>>>(sc, attb);
  out_kernel<<<BT_ / 64, 256, 0, stream>>>(attb, v2t, (float*)d_out);
}
